// MoELayer_18459769438758
// MI455X (gfx1250) — compile-verified, matrix operand order checked by reading, not run on hardware
//
#include <hip/hip_runtime.h>

// ---------------- types ----------------
typedef __bf16 v16bf __attribute__((ext_vector_type(16)));
typedef __bf16 v8bf  __attribute__((ext_vector_type(8)));
typedef __bf16 v4bf  __attribute__((ext_vector_type(4)));
typedef float  v8f   __attribute__((ext_vector_type(8)));
typedef float  v4f   __attribute__((ext_vector_type(4)));

static constexpr int NTOK = 4096;
static constexpr int DDIM = 1024;
static constexpr int HDIM = 4096;
static constexpr int NEXP = 8;

// ---------------- x -> bf16 ----------------
__global__ __launch_bounds__(256) void cvt_x_kernel(const float* __restrict__ x,
                                                    __bf16* __restrict__ xb) {
  int i = (blockIdx.x * 256 + threadIdx.x) * 4;
  v4f v = *reinterpret_cast<const v4f*>(x + i);
  v4bf o;
#pragma unroll
  for (int j = 0; j < 4; ++j) o[j] = (__bf16)v[j];
  *reinterpret_cast<v4bf*>(xb + i) = o;
}

// ---------------- gating: softmax + top-2 routing ----------------
__global__ __launch_bounds__(256) void gate_kernel(const float* __restrict__ x,
                                                   const float* __restrict__ Wg,
                                                   const float* __restrict__ bg,
                                                   float* __restrict__ scores_out,
                                                   int* __restrict__ counts,
                                                   int* __restrict__ ltok,
                                                   float* __restrict__ lw) {
  const int lane = threadIdx.x & 31;
  const int n = blockIdx.x * 8 + (threadIdx.x >> 5);  // one wave32 per token

  float acc[NEXP];
#pragma unroll
  for (int e = 0; e < NEXP; ++e) acc[e] = 0.f;

  const float* xrow = x + (size_t)n * DDIM;
  for (int d = lane; d < DDIM; d += 32) {
    float xv = xrow[d];
    v4f w0 = *reinterpret_cast<const v4f*>(Wg + (size_t)d * NEXP);
    v4f w1 = *reinterpret_cast<const v4f*>(Wg + (size_t)d * NEXP + 4);
#pragma unroll
    for (int j = 0; j < 4; ++j) {
      acc[j]     += xv * w0[j];
      acc[4 + j] += xv * w1[j];
    }
  }
#pragma unroll
  for (int off = 16; off > 0; off >>= 1) {
#pragma unroll
    for (int e = 0; e < NEXP; ++e) acc[e] += __shfl_xor(acc[e], off, 32);
  }
#pragma unroll
  for (int e = 0; e < NEXP; ++e) acc[e] += bg[e];

  float m = acc[0];
#pragma unroll
  for (int e = 1; e < NEXP; ++e) m = fmaxf(m, acc[e]);
  float s = 0.f;
  float sc[NEXP];
#pragma unroll
  for (int e = 0; e < NEXP; ++e) { sc[e] = __expf(acc[e] - m); s += sc[e]; }
  float inv = 1.0f / s;
#pragma unroll
  for (int e = 0; e < NEXP; ++e) sc[e] *= inv;

  int i1 = 0; float p1 = sc[0];
#pragma unroll
  for (int e = 1; e < NEXP; ++e) if (sc[e] > p1) { p1 = sc[e]; i1 = e; }
  int i2 = -1; float p2 = -1.f;
#pragma unroll
  for (int e = 0; e < NEXP; ++e) if (e != i1 && sc[e] > p2) { p2 = sc[e]; i2 = e; }

  if (lane == 0) {
#pragma unroll
    for (int e = 0; e < NEXP; ++e) scores_out[(size_t)n * NEXP + e] = sc[e];
    int pos1 = atomicAdd(&counts[i1], 1);
    ltok[i1 * NTOK + pos1] = n;
    lw[i1 * NTOK + pos1] = p1;
    int pos2 = atomicAdd(&counts[i2], 1);
    ltok[i2 * NTOK + pos2] = n;
    lw[i2 * NTOK + pos2] = p2;
  }
}

// ---------------- WMMA grouped GEMM ----------------
// PHASE 1: A = gathered xb rows [cnt, K=D] bf16, B = W1[e] f32 [D,H] -> hbuf = relu(A*B+b1) bf16
// PHASE 2: A = hbuf rows [cnt, K=H] bf16,       B = W2[e] f32 [H,D] -> out[tok] += w*(A*B+b2)
// Block: 256 threads (8 waves, 4x2). Block tile: 128(M) x 128(N), K-step 32.
// Wave tile: 32(M) x 64(N) = 2x4 WMMA accumulators. LDS double-buffered.
template <int PHASE>
__global__ __launch_bounds__(256) void moe_gemm(const __bf16* __restrict__ Abase,
                                                const float* __restrict__ Bbase,
                                                const float* __restrict__ bias,
                                                const int* __restrict__ cnt_ptr,
                                                const int* __restrict__ ltok,
                                                const float* __restrict__ lw,
                                                __bf16* __restrict__ hbuf,
                                                float* __restrict__ out,
                                                int K, int ldb, int Ald) {
  const int cnt = *cnt_ptr;
  const int rowbase = blockIdx.y * 128;
  if (rowbase >= cnt) return;
  const int n0 = blockIdx.x * 128;

  // LDS: pitch 40 bf16 (80B) staggers banks; all fragment chunks 16B-aligned.
  __shared__ __bf16 sA[2][128 * 40];  // A: [m][k]
  __shared__ __bf16 sB[2][128 * 40];  // B: [n][k] (transposed at staging)

  const int tid = threadIdx.x;
  const int lane = tid & 31;
  const int lanelo = lane & 15;
  const int hi = lane >> 4;
  const int w = tid >> 5;
  const int wm = w & 3;   // M sub-tile of 32
  const int wn = w >> 2;  // N sub-tile of 64

  // A staging: 2 threads per row, 32B each
  const int ar = tid >> 1;
  const int aq = tid & 1;
  const __bf16* arow;
  if (PHASE == 1) {
    int row = rowbase + ar;
    int tok = (row < cnt) ? ltok[row] : 0;
    arow = Abase + (size_t)tok * Ald;
  } else {
    arow = Abase + (size_t)(rowbase + ar) * Ald;
  }
  // B staging: each thread converts 16 f32 of one k-row
  const int bk = tid >> 3;
  const int bn = (tid & 7) * 16;

  v8f acc[2][4] = {};

  auto stage = [&](int buf, int k0) {
    // A tile (bf16 copy, 128 x 32)
    {
      const v4f* src = reinterpret_cast<const v4f*>(arow + k0 + aq * 16);
      __bf16* dst = &sA[buf][ar * 40 + aq * 16];
      *reinterpret_cast<v4f*>(dst) = src[0];
      *reinterpret_cast<v4f*>(dst + 8) = src[1];
    }
    // B tile (f32 -> bf16, transpose to [n][k], 32 x 128)
    {
      const float* src = Bbase + (size_t)(k0 + bk) * ldb + n0 + bn;
      v4f f0 = *reinterpret_cast<const v4f*>(src);
      v4f f1 = *reinterpret_cast<const v4f*>(src + 4);
      v4f f2 = *reinterpret_cast<const v4f*>(src + 8);
      v4f f3 = *reinterpret_cast<const v4f*>(src + 12);
#pragma unroll
      for (int j = 0; j < 4; ++j) {
        sB[buf][(bn + j) * 40 + bk]      = (__bf16)f0[j];
        sB[buf][(bn + 4 + j) * 40 + bk]  = (__bf16)f1[j];
        sB[buf][(bn + 8 + j) * 40 + bk]  = (__bf16)f2[j];
        sB[buf][(bn + 12 + j) * 40 + bk] = (__bf16)f3[j];
      }
    }
  };

  auto load_frag = [&](const __bf16* base, int rowIdx) -> v16bf {
    const __bf16* p = base + rowIdx * 40 + hi * 8;
    v8bf lo = *reinterpret_cast<const v8bf*>(p);
    v8bf h8 = *reinterpret_cast<const v8bf*>(p + 16);
    return __builtin_shufflevector(lo, h8, 0, 1, 2, 3, 4, 5, 6, 7, 8, 9, 10, 11,
                                   12, 13, 14, 15);
  };

  const int nk = K / 32;
  stage(0, 0);
  __syncthreads();
  for (int i = 0; i < nk; ++i) {
    const int cur = i & 1;
    // prefetch K-tile i+2 (L2-resident weights) — lowers to global_prefetch_b8
    if (i + 2 < nk) {
      __builtin_prefetch(Bbase + (size_t)((i + 2) * 32 + bk) * ldb + n0 + bn, 0, 1);
      __builtin_prefetch(arow + (i + 2) * 32 + aq * 16, 0, 1);
    }
    if (i + 1 < nk) stage(1 - cur, (i + 1) * 32);

    v16bf afrag[2], bfrag[4];
#pragma unroll
    for (int mi = 0; mi < 2; ++mi)
      afrag[mi] = load_frag(sA[cur], wm * 32 + mi * 16 + lanelo);
#pragma unroll
    for (int ni = 0; ni < 4; ++ni)
      bfrag[ni] = load_frag(sB[cur], wn * 64 + ni * 16 + lanelo);
#pragma unroll
    for (int mi = 0; mi < 2; ++mi)
#pragma unroll
      for (int ni = 0; ni < 4; ++ni)
        acc[mi][ni] = __builtin_amdgcn_wmma_f32_16x16x32_bf16(
            false, afrag[mi], false, bfrag[ni], (short)0, acc[mi][ni], false,
            false);
    __syncthreads();
  }

  // ---- epilogue ----
  // C layout: VGPR v holds (M = v + 8*hi, N = lane&15).
  // Loop (mi, v) outer -> one base address per output row; ni columns are
  // +32B/+64B immediates folded into the store/load offsets.
  const int colbase = n0 + wn * 64 + lanelo;
  float bv[4];
#pragma unroll
  for (int ni = 0; ni < 4; ++ni) bv[ni] = bias[colbase + ni * 16];

  if (rowbase + 128 <= cnt) {  // fast path: full tile, no predication
#pragma unroll
    for (int mi = 0; mi < 2; ++mi) {
#pragma unroll
      for (int v = 0; v < 8; ++v) {
        int row = rowbase + wm * 32 + mi * 16 + v + hi * 8;
        if (PHASE == 1) {
          __bf16* base = hbuf + (size_t)row * HDIM + colbase;
#pragma unroll
          for (int ni = 0; ni < 4; ++ni)
            base[ni * 16] = (__bf16)fmaxf(acc[mi][ni][v] + bv[ni], 0.f);
        } else {
          int tok = ltok[row];
          float wgt = lw[row];
          float* base = out + (size_t)tok * DDIM + colbase;
#pragma unroll
          for (int ni = 0; ni < 4; ++ni)
            base[ni * 16] += wgt * (acc[mi][ni][v] + bv[ni]);
        }
      }
    }
  } else {  // boundary tile only
#pragma unroll
    for (int mi = 0; mi < 2; ++mi) {
#pragma unroll
      for (int v = 0; v < 8; ++v) {
        int row = rowbase + wm * 32 + mi * 16 + v + hi * 8;
        if (row < cnt) {
          if (PHASE == 1) {
            __bf16* base = hbuf + (size_t)row * HDIM + colbase;
#pragma unroll
            for (int ni = 0; ni < 4; ++ni)
              base[ni * 16] = (__bf16)fmaxf(acc[mi][ni][v] + bv[ni], 0.f);
          } else {
            int tok = ltok[row];
            float wgt = lw[row];
            float* base = out + (size_t)tok * DDIM + colbase;
#pragma unroll
            for (int ni = 0; ni < 4; ++ni)
              base[ni * 16] += wgt * (acc[mi][ni][v] + bv[ni]);
          }
        }
      }
    }
  }
}

// ---------------- host launch ----------------
extern "C" void kernel_launch(void* const* d_in, const int* in_sizes, int n_in,
                              void* d_out, int out_size, void* d_ws,
                              size_t ws_size, hipStream_t stream) {
  const float* x  = (const float*)d_in[0];
  const float* W1 = (const float*)d_in[1];
  const float* b1 = (const float*)d_in[2];
  const float* W2 = (const float*)d_in[3];
  const float* b2 = (const float*)d_in[4];
  const float* Wg = (const float*)d_in[5];
  const float* bg = (const float*)d_in[6];
  float* out = (float*)d_out;
  float* scores = out + (size_t)NTOK * DDIM;

  // workspace layout (~41 MB)
  char* p = (char*)d_ws;
  __bf16* xb = (__bf16*)p;    p += (size_t)NTOK * DDIM * sizeof(__bf16);
  __bf16* hbuf = (__bf16*)p;  p += (size_t)NTOK * HDIM * sizeof(__bf16);
  int* counts = (int*)p;      p += 256;
  int* ltok = (int*)p;        p += (size_t)NEXP * NTOK * sizeof(int);
  float* lw = (float*)p;

  hipMemsetAsync(out, 0, (size_t)NTOK * DDIM * sizeof(float), stream);
  hipMemsetAsync(counts, 0, NEXP * sizeof(int), stream);

  cvt_x_kernel<<<(NTOK * DDIM) / (256 * 4), 256, 0, stream>>>(x, xb);
  gate_kernel<<<NTOK / 8, 256, 0, stream>>>(x, Wg, bg, scores, counts, ltok, lw);

  for (int e = 0; e < NEXP; ++e) {
    moe_gemm<1><<<dim3(HDIM / 128, NTOK / 128), 256, 0, stream>>>(
        xb, W1 + (size_t)e * DDIM * HDIM, b1 + (size_t)e * HDIM, counts + e,
        ltok + (size_t)e * NTOK, lw + (size_t)e * NTOK, hbuf, nullptr, DDIM,
        HDIM, DDIM);
    moe_gemm<2><<<dim3(DDIM / 128, NTOK / 128), 256, 0, stream>>>(
        hbuf, W2 + (size_t)e * HDIM * DDIM, b2 + (size_t)e * DDIM, counts + e,
        ltok + (size_t)e * NTOK, lw + (size_t)e * NTOK, nullptr, out, HDIM,
        DDIM, HDIM);
  }
}
